// m_cconv3d_20804821582125
// MI455X (gfx1250) — compile-verified, matrix operand order checked by reading, not run on hardware
//
#include <hip/hip_runtime.h>
#include <hip/hip_bf16.h>

// ---------------------------------------------------------------------------
// Sizes from the reference
// ---------------------------------------------------------------------------
#define F_BINS 512
#define T_BINS 256
#define TF     (T_BINS * F_BINS)      // 131072 elements per (b,c,d) slice
#define NSLICE 256                    // B*CPLX*D = 8*2*16
#define NPAIR  128                    // B*D complex pairs
#define CSTR   (16 * TF)              // stride from re plane to im plane (elements)

typedef _Float16 v16h __attribute__((ext_vector_type(16)));
typedef _Float16 v8h  __attribute__((ext_vector_type(8)));
typedef float    v8f  __attribute__((ext_vector_type(8)));

static __device__ __forceinline__ v8f wmma16(v16h a, v16h b, v8f c) {
    // D = A(16x32 f16) * B(32x16 f16) + C(16x16 f32)
    return __builtin_amdgcn_wmma_f32_16x16x32_f16(
        false, a, false, b, (short)0, c, false, false);
}

// Load one 16-bit WMMA A/B fragment.  Per-lane layout (wave32):
//   row/col  = lane & 15
//   k chunks = [kbase .. kbase+7] and [kbase+16 .. kbase+23], kbase = k0 + (lane>>4)*8
// Source must be "rows contiguous in k" (row-major M x K with ld = K stride).
static __device__ __forceinline__ v16h load_frag(const _Float16* p, int ld, int row, int kbase) {
    const _Float16* q = p + (size_t)row * ld + kbase;
    v8h lo = *(const v8h*)(q);
    v8h hi = *(const v8h*)(q + 16);
    v16h f;
#pragma unroll
    for (int i = 0; i < 8; ++i) { f[i] = lo[i]; f[8 + i] = hi[i]; }
    return f;
}

// ---------------------------------------------------------------------------
// Kernel 1: build f16 DFT matrices (exact periodic phase reduction) + cast W1/W2
// ---------------------------------------------------------------------------
__global__ __launch_bounds__(256) void k_consts(const float* __restrict__ W1,
                                                const float* __restrict__ W2,
                                                _Float16* __restrict__ COSh,
                                                _Float16* __restrict__ SINh,
                                                _Float16* __restrict__ W1h,
                                                _Float16* __restrict__ W2h) {
    int i = blockIdx.x * 256 + threadIdx.x;          // 0 .. 512*512-1
    int r = i >> 9, c = i & 511;
    int p = (r * c) & 511;                            // phase mod 2*pi, exact
    float ph = -6.28318530717958647692f * (float)p * (1.0f / 512.0f);
    float s, co;
    __sincosf(ph, &s, &co);
    COSh[i] = (_Float16)co;
    SINh[i] = (_Float16)s;
    W1h[i]  = (_Float16)W1[i];
    W2h[i]  = (_Float16)W2[i];
}

// ---------------------------------------------------------------------------
// Kernel 2: h[b,c,d,t,f] = (f16) x[b,c,d,f,t]   (tiled 32x32 LDS transpose)
// ---------------------------------------------------------------------------
__global__ __launch_bounds__(256) void k_transpose(const float* __restrict__ x,
                                                   _Float16* __restrict__ h) {
    __shared__ float tile[32][33];
    int bid = blockIdx.x;
    int ti  = bid & 7;           // t-tile   (8  * 32 = 256)
    int fi  = (bid >> 3) & 15;   // f-tile   (16 * 32 = 512)
    int sl  = bid >> 7;          // slice    (256)
    const float* xs = x + (size_t)sl * TF;   // [F][T] layout
    _Float16*    hs = h + (size_t)sl * TF;   // [T][F] layout
    int tx = threadIdx.x & 31, ty = threadIdx.x >> 5;   // 32 x 8
    int f0 = fi * 32, t0 = ti * 32;
#pragma unroll
    for (int j = 0; j < 4; ++j)
        tile[ty + 8 * j][tx] = xs[(size_t)(f0 + ty + 8 * j) * T_BINS + t0 + tx];
    __syncthreads();
#pragma unroll
    for (int j = 0; j < 4; ++j)
        hs[(size_t)(t0 + ty + 8 * j) * F_BINS + f0 + tx] = (_Float16)tile[tx][ty + 8 * j];
}

// ---------------------------------------------------------------------------
// Kernel 3: fused  w = softmax( (h @ W1^T + b1) @ W2^T + b2 )  per 16-row block
//   256 threads = 8 waves; wave w covers 64 columns (4 WMMA n-tiles).
//   Both GEMMs use explicit modulo-2 double buffering (no register copies).
// ---------------------------------------------------------------------------
__global__ __launch_bounds__(256) void k_mlp(const _Float16* __restrict__ h,
                                             const _Float16* __restrict__ W1h,
                                             const _Float16* __restrict__ W2h,
                                             const float* __restrict__ b1,
                                             const float* __restrict__ b2,
                                             _Float16* __restrict__ wout) {
    __shared__ __align__(16) float sbuf[16 * F_BINS];   // 32 KB
    __shared__ float red[16 * 16];
    __shared__ float rowv[16];
    _Float16* sh = (_Float16*)sbuf;                     // first 16 KB as f16

    int bid = blockIdx.x;
    int rb  = bid & 15;           // row block (16 of them -> 256 rows)
    int sl  = bid >> 4;           // slice (256)
    const _Float16* A = h + (size_t)sl * TF + (size_t)rb * 16 * F_BINS;
    _Float16* wdst    = wout + (size_t)sl * TF + (size_t)rb * 16 * F_BINS;

    int tid = threadIdx.x, wv = tid >> 5, lane = tid & 31;
    int lrow = lane & 15, khalf = (lane >> 4) * 8, rhi = khalf;

    v8f zero = {0, 0, 0, 0, 0, 0, 0, 0};

    // ---------------- GEMM1: h @ W1^T (double buffered) ----------------
    v8f acc[4] = {zero, zero, zero, zero};
    {
        v16h a0, a1, b0[4], b1[4];
        auto LD = [&](v16h& aa, v16h bb[4], int k0) {
            aa = load_frag(A, F_BINS, lrow, k0 + khalf);
#pragma unroll
            for (int t = 0; t < 4; ++t)
                bb[t] = load_frag(W1h, F_BINS, wv * 64 + t * 16 + lrow, k0 + khalf);
        };
        auto MM = [&](const v16h& aa, const v16h bb[4]) {
#pragma unroll
            for (int t = 0; t < 4; ++t) acc[t] = wmma16(aa, bb[t], acc[t]);
        };
        LD(a0, b0, 0);
#pragma unroll 1
        for (int k0 = 0; k0 < F_BINS - 64; k0 += 64) {
            LD(a1, b1, k0 + 32);
            MM(a0, b0);
            LD(a0, b0, k0 + 64);
            MM(a1, b1);
        }
        LD(a1, b1, F_BINS - 32);
        MM(a0, b0);
        MM(a1, b1);
    }
#pragma unroll
    for (int t = 0; t < 4; ++t) {
        int n = wv * 64 + t * 16 + lrow;
        float bb = b1[n];
#pragma unroll
        for (int i = 0; i < 8; ++i)
            sh[(size_t)(rhi + i) * F_BINS + n] = (_Float16)(acc[t][i] + bb);
    }
    __syncthreads();

    // ---------------- GEMM2: w1 @ W2^T (A from LDS, double buffered) ----------------
    v8f acc2[4] = {zero, zero, zero, zero};
    {
        v16h a0, a1, b0[4], b1[4];
        auto LD = [&](v16h& aa, v16h bb[4], int k0) {
            aa = load_frag(sh, F_BINS, lrow, k0 + khalf);
#pragma unroll
            for (int t = 0; t < 4; ++t)
                bb[t] = load_frag(W2h, F_BINS, wv * 64 + t * 16 + lrow, k0 + khalf);
        };
        auto MM = [&](const v16h& aa, const v16h bb[4]) {
#pragma unroll
            for (int t = 0; t < 4; ++t) acc2[t] = wmma16(aa, bb[t], acc2[t]);
        };
        LD(a0, b0, 0);
#pragma unroll 1
        for (int k0 = 0; k0 < F_BINS - 64; k0 += 64) {
            LD(a1, b1, k0 + 32);
            MM(a0, b0);
            LD(a0, b0, k0 + 64);
            MM(a1, b1);
        }
        LD(a1, b1, F_BINS - 32);
        MM(a0, b0);
        MM(a1, b1);
    }
    __syncthreads();              // all GEMM2 reads of sh done; reuse as f32
#pragma unroll
    for (int t = 0; t < 4; ++t) {
        int n = wv * 64 + t * 16 + lrow;
        float bb = b2[n];
#pragma unroll
        for (int i = 0; i < 8; ++i)
            sbuf[(size_t)(rhi + i) * F_BINS + n] = acc2[t][i] + bb;
    }
    __syncthreads();

    // softmax over 512 columns per row; 16 threads per row, 32 cols each
    int row = tid >> 4, seg = tid & 15;
    float m = -1e30f;
    for (int c = 0; c < 32; ++c)
        m = fmaxf(m, sbuf[row * F_BINS + seg * 32 + c]);
    red[row * 16 + seg] = m;
    __syncthreads();
    if (seg == 0) {
        float mm = red[row * 16];
        for (int c = 1; c < 16; ++c) mm = fmaxf(mm, red[row * 16 + c]);
        rowv[row] = mm;
    }
    __syncthreads();
    float rm = rowv[row];
    float ssum = 0.0f;
    for (int c = 0; c < 32; ++c) {
        int idx = row * F_BINS + seg * 32 + c;
        float e = __expf(sbuf[idx] - rm);
        sbuf[idx] = e;
        ssum += e;
    }
    red[row * 16 + seg] = ssum;
    __syncthreads();
    if (seg == 0) {
        float ss = 0.0f;
        for (int c = 0; c < 16; ++c) ss += red[row * 16 + c];
        rowv[row] = ss;
    }
    __syncthreads();
    float inv = 1.0f / rowv[row];
    for (int c = 0; c < 32; ++c) {
        int idx = row * F_BINS + seg * 32 + c;
        wdst[idx] = (_Float16)(sbuf[idx] * inv);
    }
}

// ---------------------------------------------------------------------------
// Kernel 4: complex iDFT.  dst_re = (re@COS + im@SIN)/F, dst_im = (im@COS - re@SIN)/F
//   COS/SIN symmetric -> B[k,n] read as row n, contiguous k.
//   256 threads = 8 waves; wave covers 32 columns (2 n-tiles x 4 accumulators).
//   Explicit modulo-2 double buffering (no register copies).
// ---------------------------------------------------------------------------
__global__ __launch_bounds__(256) void k_idft(const _Float16* __restrict__ src,
                                              _Float16* __restrict__ dst,
                                              const _Float16* __restrict__ COSh,
                                              const _Float16* __restrict__ SINh) {
    int bid  = blockIdx.x;
    int half = bid & 1;              // which 256-column half
    int rb   = (bid >> 1) & 15;      // 16 row blocks of 16 -> 256 rows
    int sp   = bid >> 5;             // (b,d) pair, 128
    int b = sp >> 4, d = sp & 15;
    size_t reOff = (size_t)(b * 32 + d) * TF;
    size_t imOff = reOff + CSTR;
    const _Float16* Are = src + reOff + (size_t)rb * 16 * F_BINS;
    const _Float16* Aim = src + imOff + (size_t)rb * 16 * F_BINS;
    _Float16* Dre = dst + reOff + (size_t)rb * 16 * F_BINS;
    _Float16* Dim = dst + imOff + (size_t)rb * 16 * F_BINS;

    int tid = threadIdx.x, wv = tid >> 5, lane = tid & 31;
    int lrow = lane & 15, khalf = (lane >> 4) * 8, rhi = khalf;
    int n0 = half * 256 + wv * 32;

    v8f zero = {0, 0, 0, 0, 0, 0, 0, 0};
    v8f rc[2] = {zero, zero}, is_[2] = {zero, zero};
    v8f ic[2] = {zero, zero}, rs[2]  = {zero, zero};

    v16h ar0, ai0, bc0[2], bs0[2];
    v16h ar1, ai1, bc1[2], bs1[2];
    auto LD = [&](v16h& arr, v16h& aii, v16h bcc[2], v16h bss[2], int k0) {
        arr = load_frag(Are, F_BINS, lrow, k0 + khalf);
        aii = load_frag(Aim, F_BINS, lrow, k0 + khalf);
#pragma unroll
        for (int t = 0; t < 2; ++t) {
            bcc[t] = load_frag(COSh, F_BINS, n0 + t * 16 + lrow, k0 + khalf);
            bss[t] = load_frag(SINh, F_BINS, n0 + t * 16 + lrow, k0 + khalf);
        }
    };
    auto MM = [&](const v16h& arr, const v16h& aii, const v16h bcc[2], const v16h bss[2]) {
#pragma unroll
        for (int t = 0; t < 2; ++t) {
            rc[t]  = wmma16(arr, bcc[t], rc[t]);
            is_[t] = wmma16(aii, bss[t], is_[t]);
            ic[t]  = wmma16(aii, bcc[t], ic[t]);
            rs[t]  = wmma16(arr, bss[t], rs[t]);
        }
    };
    LD(ar0, ai0, bc0, bs0, 0);
#pragma unroll 1
    for (int k0 = 0; k0 < F_BINS - 64; k0 += 64) {
        LD(ar1, ai1, bc1, bs1, k0 + 32);
        MM(ar0, ai0, bc0, bs0);
        LD(ar0, ai0, bc0, bs0, k0 + 64);
        MM(ar1, ai1, bc1, bs1);
    }
    LD(ar1, ai1, bc1, bs1, F_BINS - 32);
    MM(ar0, ai0, bc0, bs0);
    MM(ar1, ai1, bc1, bs1);

    const float sc = 1.0f / 512.0f;
#pragma unroll
    for (int t = 0; t < 2; ++t) {
        int n = n0 + t * 16 + lrow;
#pragma unroll
        for (int i = 0; i < 8; ++i) {
            size_t o = (size_t)(rhi + i) * F_BINS + n;
            Dre[o] = (_Float16)((rc[t][i] + is_[t][i]) * sc);
            Dim[o] = (_Float16)((ic[t][i] - rs[t][i]) * sc);
        }
    }
}

// ---------------------------------------------------------------------------
// Kernel 5: elementwise complex multiply, in place into a (8 f16 / thread, b128)
//   pr = ar*br - ai*bi ; pi = ar*bi + ai*br
// ---------------------------------------------------------------------------
__global__ __launch_bounds__(256) void k_cmul(_Float16* __restrict__ a,
                                              const _Float16* __restrict__ bm) {
    size_t g = (size_t)blockIdx.x * 256 + threadIdx.x;   // groups of 8 elements
    int bd  = (int)(g >> 14);            // TF/8 = 16384 groups per (b,d) slice
    int pos = (int)(g & 16383) * 8;
    int b = bd >> 4, d = bd & 15;
    size_t reI = (size_t)(b * 32 + d) * TF + pos;
    size_t imI = reI + CSTR;
    v8h xr = *(const v8h*)(a + reI),  xi = *(const v8h*)(a + imI);
    v8h yr = *(const v8h*)(bm + reI), yi = *(const v8h*)(bm + imI);
    v8h pr, pi;
#pragma unroll
    for (int i = 0; i < 8; ++i) {
        float ar = (float)xr[i], ai = (float)xi[i];
        float br = (float)yr[i], bi = (float)yi[i];
        pr[i] = (_Float16)(ar * br - ai * bi);
        pi[i] = (_Float16)(ar * bi + ai * br);
    }
    *(v8h*)(a + reI) = pr;
    *(v8h*)(a + imI) = pi;
}

// ---------------------------------------------------------------------------
// Kernel 6: forward DFT * F with free transpose:
//   out'[f,t] = F * ( COS[f,:]·re[t,:] - SIN[f,:]·im[t,:] )   (re plane)
//   out'[f,t] = F * ( SIN[f,:]·re[t,:] + COS[f,:]·im[t,:] )   (im plane)
//   A = COS/SIN rows f (symmetric), B[k,t] = prod[t,k] rows t contiguous k.
//   Coalesced f32 stores along t.   Explicit modulo-2 double buffering.
// ---------------------------------------------------------------------------
__global__ __launch_bounds__(256) void k_dft_t(const _Float16* __restrict__ prod,
                                               float* __restrict__ out,
                                               const _Float16* __restrict__ COSh,
                                               const _Float16* __restrict__ SINh) {
    int bid = blockIdx.x;
    int fb  = bid & 31;              // 32 f blocks of 16 -> 512
    int sp  = bid >> 5;              // (b,d) pair, 128
    int b = sp >> 4, d = sp & 15;
    size_t reOff = (size_t)(b * 32 + d) * TF;
    size_t imOff = reOff + CSTR;
    const _Float16* Pre = prod + reOff;
    const _Float16* Pim = prod + imOff;
    float* Ore = out + reOff;        // out slice layout [F][T]
    float* Oim = out + imOff;

    int tid = threadIdx.x, wv = tid >> 5, lane = tid & 31;
    int lrow = lane & 15, khalf = (lane >> 4) * 8, rhi = khalf;
    int f0 = fb * 16, t0 = wv * 32;

    v8f zero = {0, 0, 0, 0, 0, 0, 0, 0};
    v8f cr[2] = {zero, zero}, si[2] = {zero, zero};
    v8f sr[2] = {zero, zero}, ci[2] = {zero, zero};

    v16h ac0, as0, br0[2], bi0[2];
    v16h ac1, as1, br1[2], bi1[2];
    auto LD = [&](v16h& acc_, v16h& ass_, v16h brr[2], v16h bii[2], int k0) {
        acc_ = load_frag(COSh, F_BINS, f0 + lrow, k0 + khalf);
        ass_ = load_frag(SINh, F_BINS, f0 + lrow, k0 + khalf);
#pragma unroll
        for (int t = 0; t < 2; ++t) {
            brr[t] = load_frag(Pre, F_BINS, t0 + t * 16 + lrow, k0 + khalf);
            bii[t] = load_frag(Pim, F_BINS, t0 + t * 16 + lrow, k0 + khalf);
        }
    };
    auto MM = [&](const v16h& acc_, const v16h& ass_, const v16h brr[2], const v16h bii[2]) {
#pragma unroll
        for (int t = 0; t < 2; ++t) {
            cr[t] = wmma16(acc_, brr[t], cr[t]);
            si[t] = wmma16(ass_, bii[t], si[t]);
            sr[t] = wmma16(ass_, brr[t], sr[t]);
            ci[t] = wmma16(acc_, bii[t], ci[t]);
        }
    };
    LD(ac0, as0, br0, bi0, 0);
#pragma unroll 1
    for (int k0 = 0; k0 < F_BINS - 64; k0 += 64) {
        LD(ac1, as1, br1, bi1, k0 + 32);
        MM(ac0, as0, br0, bi0);
        LD(ac0, as0, br0, bi0, k0 + 64);
        MM(ac1, as1, br1, bi1);
    }
    LD(ac1, as1, br1, bi1, F_BINS - 32);
    MM(ac0, as0, br0, bi0);
    MM(ac1, as1, br1, bi1);

    const float sc = 512.0f;
#pragma unroll
    for (int t = 0; t < 2; ++t) {
        int tt = t0 + t * 16 + lrow;
#pragma unroll
        for (int i = 0; i < 8; ++i) {
            int f = f0 + rhi + i;
            size_t o = (size_t)f * T_BINS + tt;
            Ore[o] = (cr[t][i] - si[t][i]) * sc;
            Oim[o] = (sr[t][i] + ci[t][i]) * sc;
        }
    }
}

// ---------------------------------------------------------------------------
// Host launcher
// ---------------------------------------------------------------------------
extern "C" void kernel_launch(void* const* d_in, const int* in_sizes, int n_in,
                              void* d_out, int out_size, void* d_ws, size_t ws_size,
                              hipStream_t stream) {
    const float* x  = (const float*)d_in[0];
    const float* W1 = (const float*)d_in[1];
    const float* b1 = (const float*)d_in[2];
    const float* W2 = (const float*)d_in[3];
    const float* b2 = (const float*)d_in[4];
    float* out = (float*)d_out;

    char* ws = (char*)d_ws;
    const size_t MAT16 = (size_t)F_BINS * F_BINS * sizeof(_Float16); // 512 KB
    const size_t BIG16 = (size_t)NSLICE * TF * sizeof(_Float16);     // 64 MB
    _Float16* COSh = (_Float16*)(ws);
    _Float16* SINh = (_Float16*)(ws + MAT16);
    _Float16* W1h  = (_Float16*)(ws + 2 * MAT16);
    _Float16* W2h  = (_Float16*)(ws + 3 * MAT16);
    _Float16* h_h  = (_Float16*)(ws + 4 * MAT16);                    // x transposed, f16
    _Float16* w_h  = (_Float16*)(ws + 4 * MAT16 + BIG16);            // softmax probs
    _Float16* xif  = (_Float16*)(ws + 4 * MAT16 + 2 * BIG16);        // idft(h) -> later prod (in place)
    _Float16* yif  = (_Float16*)(ws + 4 * MAT16 + 3 * BIG16);        // idft(w)

    // 1) DFT matrices + weight casts
    k_consts<<<(F_BINS * F_BINS) / 256, 256, 0, stream>>>(W1, W2, COSh, SINh, W1h, W2h);
    // 2) transpose + cast x -> h
    k_transpose<<<NSLICE * 16 * 8, 256, 0, stream>>>(x, h_h);
    // 3) fused MLP + softmax -> w
    k_mlp<<<NSLICE * 16, 256, 0, stream>>>(h_h, W1h, W2h, b1, b2, w_h);
    // 4) iDFT of h and of w  (2 column-halves per row block)
    k_idft<<<NPAIR * 32, 256, 0, stream>>>(h_h, xif, COSh, SINh);
    k_idft<<<NPAIR * 32, 256, 0, stream>>>(w_h, yif, COSh, SINh);
    // 5) complex multiply (in place into xif), 8 f16 per thread
    k_cmul<<<(NPAIR * TF / 8) / 256, 256, 0, stream>>>(xif, yif);
    // 6) forward DFT * F with transposed, coalesced store into d_out
    k_dft_t<<<NPAIR * 32, 256, 0, stream>>>(xif, out, COSh, SINh);

    (void)in_sizes; (void)n_in; (void)out_size; (void)ws_size;
}
